// VFunction_43825846288948
// MI455X (gfx1250) — matrix operand order repaired and checked by reading, not compiled, not run on hardware
//
#include <hip/hip_runtime.h>

typedef __bf16 bf16;
typedef __bf16 v16bf __attribute__((ext_vector_type(16)));
typedef float  v8f   __attribute__((ext_vector_type(8)));
typedef float  v4f   __attribute__((ext_vector_type(4)));
typedef unsigned int v4u __attribute__((ext_vector_type(4)));

namespace {
constexpr int NB = 1024, NT = 16, ND = 5120, NH = 1024;
constexpr int NM = NB * NT;                 // 16384 rows (B*T)
constexpr float LAM = 0.95f;
constexpr float DISCOUNT = 1.0f - 1.0f / 333.0f;
}

union Frag { v16bf v; v4u u[2]; };

__device__ __forceinline__ float silu_f(float x) {
  return x / (1.0f + __expf(-x));
}

// ---- CDNA5 async global->LDS copy (16B), tracked by ASYNCcnt ----
__device__ __forceinline__ void async_copy_b128(const void* lds_dst, const void* gptr) {
  unsigned ldsOff = (unsigned)(unsigned long long)lds_dst;   // LDS aperture: low 32 bits = LDS offset
  asm volatile("global_load_async_to_lds_b128 %0, %1, off"
               :: "v"(ldsOff), "v"(gptr) : "memory");
}
__device__ __forceinline__ void wait_async_le12() {
  asm volatile("s_wait_asynccnt 0xc" ::: "memory");
}
__device__ __forceinline__ void wait_async_0() {
  asm volatile("s_wait_asynccnt 0x0" ::: "memory");
}

// ---------- feat f32 -> bf16 (one pass; kills 8x redundant cvt + halves re-read bytes) ----------
__global__ void cvt_bf16_kernel(const float* __restrict__ in, bf16* __restrict__ out, long n) {
  long i = ((long)blockIdx.x * blockDim.x + threadIdx.x) * 8;
  long stride = (long)gridDim.x * blockDim.x * 8;
  for (; i < n; i += stride) {
    v4f x0 = *(const v4f*)(in + i);
    v4f x1 = *(const v4f*)(in + i + 4);
    union { bf16 h[8]; v4u u; } pk;
    pk.h[0] = (bf16)x0.x; pk.h[1] = (bf16)x0.y; pk.h[2] = (bf16)x0.z; pk.h[3] = (bf16)x0.w;
    pk.h[4] = (bf16)x1.x; pk.h[5] = (bf16)x1.y; pk.h[6] = (bf16)x1.z; pk.h[7] = (bf16)x1.w;
    *(v4u*)(out + i) = pk.u;
  }
}

// ---------- weight convert + transpose: in f32 [R][C] -> out bf16 [C][R] ----------
__global__ void cvt_transpose_kernel(const float* __restrict__ in, bf16* __restrict__ out,
                                     int R, int C) {
  long total = (long)R * C;
  for (long idx = (long)blockIdx.x * blockDim.x + threadIdx.x; idx < total;
       idx += (long)gridDim.x * blockDim.x) {
    int r = (int)(idx / C), c = (int)(idx % C);
    out[(long)c * R + r] = (bf16)in[idx];
  }
}

__global__ void zero_kernel(float* __restrict__ p, int n) {
  int i = blockIdx.x * blockDim.x + threadIdx.x;
  if (i < n) p[i] = 0.0f;
}

// ---------- double-buffered async bf16 WMMA GEMM, 128x256 block tile ----------
// C[M,1024] = silu(A[M,K] * W[K,1024] + bias);  W pre-transposed Wt[n][k].
// FUSE: accumulate sum_n silu(..)*w3[n] into dotOut[z][m] via atomics instead of storing C.
template<int K, bool FUSE>
__global__ __launch_bounds__(256)
void gemm_silu_kernel(const bf16* __restrict__ Ab, long aStrideZ,
                      const bf16* __restrict__ Wt, long wStrideZ,
                      const float* __restrict__ bias0, const float* __restrict__ bias1,
                      bf16* __restrict__ Cout, long cStrideZ,
                      const float* __restrict__ w3n, const float* __restrict__ w3s,
                      float* __restrict__ dotOut) {
  const int z  = blockIdx.z;               // 0 = net, 1 = slow
  const int n0 = blockIdx.x * 256;
  const int m0 = blockIdx.y * 128;
  const int tid  = threadIdx.x;
  const int lane = tid & 31, wid = tid >> 5;
  const int waveM = wid & 1, waveN = wid >> 1;   // 2 x 4 wave grid -> 64x64 per wave

  __shared__ bf16 ldsA[2][128][72];
  __shared__ bf16 ldsB[2][256][72];

  const bf16* A  = Ab + (long)z * aStrideZ;
  const bf16* Wz = Wt + (long)z * wStrideZ;
  const float* biasz = z ? bias1 : bias0;

  // staging: A tile 128x64 = 1024 16B chunks, B tile 256x64 = 2048 chunks; 12 asyncs/thread
  auto stage = [&](int buf, int k0) {
#pragma unroll
    for (int i = 0; i < 4; ++i) {
      int idx = tid + i * 256;                     // 0..1023
      int row = idx >> 3, kv = (idx & 7) << 3;     // kv in bf16 elements (8 per 16B)
      async_copy_b128(&ldsA[buf][row][kv], A + (long)(m0 + row) * K + (k0 + kv));
    }
#pragma unroll
    for (int i = 0; i < 8; ++i) {
      int idx = tid + i * 256;                     // 0..2047
      int row = idx >> 3, kv = (idx & 7) << 3;
      async_copy_b128(&ldsB[buf][row][kv], Wz + (long)(n0 + row) * K + (k0 + kv));
    }
  };

  v8f acc[4][4];
  const v8f zero8 = {0.f, 0.f, 0.f, 0.f, 0.f, 0.f, 0.f, 0.f};
#pragma unroll
  for (int i = 0; i < 4; ++i)
#pragma unroll
    for (int j = 0; j < 4; ++j) acc[i][j] = zero8;

  stage(0, 0);
  int cur = 0;
  for (int k0 = 0; k0 < K; k0 += 64) {
    if (k0 + 64 < K) {
      stage(cur ^ 1, k0 + 64);    // prefetch next tile into other buffer
      wait_async_le12();          // 12 outstanding = just-issued ones -> current tile complete
    } else {
      wait_async_0();
    }
    __syncthreads();

#pragma unroll
    for (int kk = 0; kk < 64; kk += 32) {
      Frag a[4], bfr[4];
      const int ar = waveM * 64 + (lane & 15);
      const int ak = kk + ((lane >> 4) << 3);        // A frag: half-wave K split (+0/+8, +16)
#pragma unroll
      for (int im = 0; im < 4; ++im) {
        const bf16* p = &ldsA[cur][ar + im * 16][ak];
        a[im].u[0] = *(const v4u*)p;
        a[im].u[1] = *(const v4u*)(p + 16);
      }
      const int br = waveN * 64 + (lane & 15);
      const int bk = kk + ((lane >> 4) << 3);        // B frag: half-wave K split (+0/+8, +16)
#pragma unroll
      for (int in_ = 0; in_ < 4; ++in_) {
        const bf16* p = &ldsB[cur][br + in_ * 16][bk];
        bfr[in_].u[0] = *(const v4u*)p;
        bfr[in_].u[1] = *(const v4u*)(p + 16);
      }
#pragma unroll
      for (int im = 0; im < 4; ++im)
#pragma unroll
        for (int in_ = 0; in_ < 4; ++in_)
          acc[im][in_] = __builtin_amdgcn_wmma_f32_16x16x32_bf16(
              false, a[im].v, false, bfr[in_].v, (short)0, acc[im][in_], false, false);
    }
    __syncthreads();              // all waves done reading buf before it is overwritten
    cur ^= 1;
  }

  // ---- epilogue ----
  if constexpr (!FUSE) {
    bf16* C = Cout + (long)z * cStrideZ;
#pragma unroll
    for (int in_ = 0; in_ < 4; ++in_) {
      int n = n0 + waveN * 64 + in_ * 16 + (lane & 15);
      float bv = biasz[n];
#pragma unroll
      for (int im = 0; im < 4; ++im) {
        int mbase = m0 + waveM * 64 + im * 16 + ((lane >> 4) << 3);
#pragma unroll
        for (int r = 0; r < 8; ++r)
          C[(long)(mbase + r) * NH + n] = (bf16)silu_f(acc[im][in_][r] + bv);
      }
    }
  } else {
    const float* w3z = z ? w3s : w3n;
    float* dz = dotOut + (long)z * NM;
#pragma unroll
    for (int im = 0; im < 4; ++im) {
      float p[8] = {0.f, 0.f, 0.f, 0.f, 0.f, 0.f, 0.f, 0.f};
#pragma unroll
      for (int in_ = 0; in_ < 4; ++in_) {
        int n = n0 + waveN * 64 + in_ * 16 + (lane & 15);
        float bv = biasz[n], wv = w3z[n];
#pragma unroll
        for (int r = 0; r < 8; ++r)
          p[r] += silu_f(acc[im][in_][r] + bv) * wv;
      }
#pragma unroll
      for (int off = 1; off < 16; off <<= 1)
#pragma unroll
        for (int r = 0; r < 8; ++r)
          p[r] += __shfl_xor(p[r], off, 32);
      if ((lane & 15) == 0) {
        int mbase = m0 + waveM * 64 + im * 16 + ((lane >> 4) << 3);
#pragma unroll
        for (int r = 0; r < 8; ++r)
          atomicAdd(&dz[mbase + r], p[r]);
      }
    }
  }
}

// ---------- lambda-return scan + output assembly (one thread per batch row) ----------
__global__ void scan_kernel(const float* __restrict__ rew, const float* __restrict__ cont,
                            const float* __restrict__ dot,    // [2][NM] raw dot (no b3)
                            const float* __restrict__ net_b3, const float* __restrict__ slow_b3,
                            const float* __restrict__ voffset, const float* __restrict__ vscale,
                            float* __restrict__ out) {
  int b = blockIdx.x * blockDim.x + threadIdx.x;
  if (b >= NB) return;
  const float vs = vscale[0], vo = voffset[0];
  const float nb3 = net_b3[0], sb3 = slow_b3[0];
  float cr[NT], sc[NT], tar[NT];
#pragma unroll
  for (int t = 0; t < NT; ++t) {
    float c = dot[b * NT + t] + nb3;
    float s = dot[NM + b * NT + t] + sb3;
    cr[t] = c; sc[t] = s; tar[t] = s * vs + vo;
  }
  float* o_rew = out;
  float* o_ret = out + NM;
  float* o_tar = o_ret + NB * (NT - 1);
  float* o_cr  = o_tar + NM;
  float* o_sc  = o_cr + NM;
  float carry = tar[NT - 1];
#pragma unroll
  for (int i = NT - 2; i >= 0; --i) {
    float d = cont[b * NT + i + 1] * DISCOUNT;
    float interm = rew[b * NT + i + 1] + (1.0f - LAM) * d * tar[i + 1];
    carry = interm + d * LAM * carry;
    o_ret[b * (NT - 1) + i] = carry;
  }
#pragma unroll
  for (int t = 0; t < NT; ++t) {
    o_rew[b * NT + t] = rew[b * NT + t];
    o_tar[b * NT + t] = tar[t];
    o_cr[b * NT + t]  = cr[t];
    o_sc[b * NT + t]  = sc[t];
  }
}

extern "C" void kernel_launch(void* const* d_in, const int* in_sizes, int n_in,
                              void* d_out, int out_size, void* d_ws, size_t ws_size,
                              hipStream_t stream) {
  (void)in_sizes; (void)n_in; (void)out_size; (void)ws_size;
  const float* feat    = (const float*)d_in[0];
  const float* rew     = (const float*)d_in[1];
  const float* cont    = (const float*)d_in[2];
  const float* net_w1  = (const float*)d_in[3];
  const float* net_b1  = (const float*)d_in[4];
  const float* net_w2  = (const float*)d_in[5];
  const float* net_b2  = (const float*)d_in[6];
  const float* net_w3  = (const float*)d_in[7];
  const float* net_b3  = (const float*)d_in[8];
  const float* slow_w1 = (const float*)d_in[9];
  const float* slow_b1 = (const float*)d_in[10];
  const float* slow_w2 = (const float*)d_in[11];
  const float* slow_b2 = (const float*)d_in[12];
  const float* slow_w3 = (const float*)d_in[13];
  const float* slow_b3 = (const float*)d_in[14];
  const float* voffset = (const float*)d_in[15];
  const float* vscale  = (const float*)d_in[16];

  // workspace layout (~260 MB)
  bf16*  featb = (bf16*)d_ws;                    // [NM][ND] feat in bf16 (L2-resident, 168 MB)
  bf16*  w1t   = featb + (long)NM * ND;          // [2][NH][ND]
  bf16*  w2t   = w1t + 2L * NH * ND;             // [2][NH][NH]
  bf16*  H1    = w2t + 2L * NH * NH;             // [2][NM][NH]
  float* dot   = (float*)(H1 + 2L * NM * NH);    // [2][NM]

  cvt_bf16_kernel<<<8192, 256, 0, stream>>>(feat, featb, (long)NM * ND);
  cvt_transpose_kernel<<<4096, 256, 0, stream>>>(net_w1,  w1t,               ND, NH);
  cvt_transpose_kernel<<<4096, 256, 0, stream>>>(slow_w1, w1t + (long)NH*ND, ND, NH);
  cvt_transpose_kernel<<<1024, 256, 0, stream>>>(net_w2,  w2t,               NH, NH);
  cvt_transpose_kernel<<<1024, 256, 0, stream>>>(slow_w2, w2t + (long)NH*NH, NH, NH);
  zero_kernel<<<(2 * NM + 255) / 256, 256, 0, stream>>>(dot, 2 * NM);

  // layer 1: [16384,5120] x [5120,1024] for both nets (grid.z)
  gemm_silu_kernel<ND, false><<<dim3(4, 128, 2), 256, 0, stream>>>(
      featb, 0, w1t, (long)NH * ND, net_b1, slow_b1,
      H1, (long)NM * NH, nullptr, nullptr, nullptr);

  // layer 2 + fused layer-3 dot: [16384,1024] x [1024,1024] -> dot[2][16384]
  gemm_silu_kernel<NH, true><<<dim3(4, 128, 2), 256, 0, stream>>>(
      H1, (long)NM * NH, w2t, (long)NH * NH, net_b2, slow_b2,
      nullptr, 0, net_w3, slow_w3, dot);

  scan_kernel<<<4, 256, 0, stream>>>(rew, cont, dot, net_b3, slow_b3,
                                     voffset, vscale, (float*)d_out);
}
